// MDTAWithStrategy_747324310133
// MI455X (gfx1250) — matrix operand order repaired and checked by reading, not compiled, not run on hardware
//
#include <hip/hip_runtime.h>
#include <hip/hip_bf16.h>

typedef __attribute__((ext_vector_type(16))) _Float16 v16h;
typedef __attribute__((ext_vector_type(8)))  _Float16 v8h;
typedef __attribute__((ext_vector_type(8)))  float    v8f;

#define DIMC   512
#define NHEAD  8
#define HDIM   64
#define SEQ_N  4096
#define SEQ_K  256
#define BATCH  8

// ---------------------------------------------------------------------------
// f32 -> f16 conversion (grid-stride)
// ---------------------------------------------------------------------------
__global__ void cvt_f32_to_f16(const float* __restrict__ in,
                               _Float16* __restrict__ out, int n) {
    int i = blockIdx.x * blockDim.x + threadIdx.x;
    int stride = gridDim.x * blockDim.x;
    for (; i < n; i += stride) out[i] = (_Float16)in[i];
}

// ---------------------------------------------------------------------------
// NT GEMM: C[M x 512] = A[M x 512] @ W^T (W is 512x512, row-major over K)
// One wave (32 threads) computes a 16 x 128 output strip using
// v_wmma_f32_16x16x32_f16. FINAL=true adds f32 residual and writes f32.
// ---------------------------------------------------------------------------
template <bool FINAL>
__global__ __launch_bounds__(32) void gemm_nt_wmma(
    const _Float16* __restrict__ A, const _Float16* __restrict__ W,
    _Float16* __restrict__ Cf16, const float* __restrict__ resid,
    float* __restrict__ Cf32) {
    const int lane = threadIdx.x & 31;
    const int lm   = lane & 15;   // row (A) / col (B) within tile
    const int hi   = lane >> 4;   // K-half selector
    const int m0   = blockIdx.x * 16;
    const int n0   = blockIdx.y * 128;

    v8f acc[8];
    const v8f zacc = {0.f, 0.f, 0.f, 0.f, 0.f, 0.f, 0.f, 0.f};
    for (int t = 0; t < 8; ++t) acc[t] = zacc;

    const _Float16* arow = A + (size_t)(m0 + lm) * DIMC;

    for (int kb = 0; kb < DIMC; kb += 32) {
        // A fragment: lane lo holds K {0..15}, lane hi {16..31}
        v8h alo = *(const v8h*)(arow + kb + hi * 16);
        v8h ahi = *(const v8h*)(arow + kb + 8 + hi * 16);
        v16h af;
        for (int j = 0; j < 8; ++j) { af[j] = alo[j]; af[8 + j] = ahi[j]; }
        for (int t = 0; t < 8; ++t) {
            // B fragment (B[k,n] = W[n,k]): lane holds col n, K hi*16 + 0..15
            const _Float16* wrow = W + (size_t)(n0 + t * 16 + lm) * DIMC;
            v16h bf = *(const v16h*)(wrow + kb + hi * 16);
            acc[t] = __builtin_amdgcn_wmma_f32_16x16x32_f16(
                false, af, false, bf, (short)0, acc[t], false, false);
        }
    }

    for (int t = 0; t < 8; ++t) {
        for (int i = 0; i < 8; ++i) {
            const size_t row = (size_t)(m0 + i + hi * 8);
            const size_t col = (size_t)(n0 + t * 16 + lm);
            if (FINAL) {
                Cf32[row * DIMC + col] = resid[row * DIMC + col] + acc[t][i];
            } else {
                Cf16[row * DIMC + col] = (_Float16)acc[t][i];
            }
        }
    }
}

// ---------------------------------------------------------------------------
// Attention: per (b, h, 64-row tile). 4 waves, each wave owns 16 query rows.
// K rows normalized + staged to LDS; V staged transposed; softmax in regs;
// P re-striped C-layout -> A-layout through wave-private LDS.
// ---------------------------------------------------------------------------
__global__ __launch_bounds__(128, 1) void attn_wmma(
    const _Float16* __restrict__ Q, const _Float16* __restrict__ Kp,
    const _Float16* __restrict__ Vp, const float* __restrict__ temperature,
    _Float16* __restrict__ O) {
    __shared__ _Float16 lds_k[SEQ_K * HDIM];     // 32 KB normalized K rows
    __shared__ _Float16 lds_vt[HDIM * SEQ_K];    // 32 KB V transposed [d][k]
    __shared__ _Float16 lds_p[4][16 * SEQ_K];    // 32 KB per-wave P tiles

    const int b    = blockIdx.z;
    const int h    = blockIdx.y;
    const int tid  = threadIdx.x;
    const int wave = tid >> 5;
    const int lane = tid & 31;
    const int lm   = lane & 15;
    const int hi   = lane >> 4;

    // ---- stage K (L2-normalized) and V^T into LDS ----
    for (int r = tid; r < SEQ_K; r += blockDim.x) {
        const _Float16* krow = Kp + ((size_t)(b * SEQ_K + r)) * DIMC + h * HDIM;
        const _Float16* vrow = Vp + ((size_t)(b * SEQ_K + r)) * DIMC + h * HDIM;
        v8h kv[8];
        float ss = 0.f;
        for (int c = 0; c < 8; ++c) {
            kv[c] = *(const v8h*)(krow + c * 8);
            for (int j = 0; j < 8; ++j) { float x = (float)kv[c][j]; ss += x * x; }
        }
        const float sc = 1.0f / fmaxf(sqrtf(ss), 1e-12f);
        for (int c = 0; c < 8; ++c) {
            v8h o;
            for (int j = 0; j < 8; ++j) o[j] = (_Float16)((float)kv[c][j] * sc);
            *(v8h*)(&lds_k[r * HDIM + c * 8]) = o;
        }
        for (int c = 0; c < 8; ++c) {
            v8h vv = *(const v8h*)(vrow + c * 8);
            for (int j = 0; j < 8; ++j) lds_vt[(c * 8 + j) * SEQ_K + r] = vv[j];
        }
    }
    __syncthreads();

    const float temp = temperature[h];
    const int   n0   = blockIdx.x * 64 + wave * 16;
    const _Float16* qrow =
        Q + ((size_t)(b * SEQ_N + n0 + lm)) * DIMC + h * HDIM;

    // ---- load + L2-normalize Q fragments (row split across lane pair m,m+16)
    v16h qa[2];
    float qss = 0.f;
    for (int c = 0; c < 2; ++c) {
        v8h lo  = *(const v8h*)(qrow + c * 32 + hi * 16);
        v8h hv  = *(const v8h*)(qrow + c * 32 + 8 + hi * 16);
        for (int j = 0; j < 8; ++j) { qa[c][j] = lo[j]; qa[c][8 + j] = hv[j]; }
        for (int j = 0; j < 16; ++j) { float x = (float)qa[c][j]; qss += x * x; }
    }
    qss += __shfl_xor(qss, 16);
    const float qsc = 1.0f / fmaxf(sqrtf(qss), 1e-12f);
    for (int c = 0; c < 2; ++c)
        for (int j = 0; j < 16; ++j)
            qa[c][j] = (_Float16)((float)qa[c][j] * qsc);

    // ---- scores: 16x256 = 16 tiles of 16x16, K-dim = 64 (2 wmma steps) ----
    v8f sacc[16];
    const v8f zacc = {0.f, 0.f, 0.f, 0.f, 0.f, 0.f, 0.f, 0.f};
    for (int t = 0; t < 16; ++t) sacc[t] = zacc;
    for (int c = 0; c < 2; ++c) {
        for (int t = 0; t < 16; ++t) {
            v16h bf = *(const v16h*)(&lds_k[(t * 16 + lm) * HDIM + c * 32 + hi * 16]);
            sacc[t] = __builtin_amdgcn_wmma_f32_16x16x32_f16(
                false, qa[c], false, bf, (short)0, sacc[t], false, false);
        }
    }

    // ---- softmax over 256 keys (16-lane shuffle reductions, rows in
    //      lo/hi lane halves reduce independently, matching C layout) ----
    for (int i = 0; i < 8; ++i) {
        float m = -3.0e38f;
        for (int t = 0; t < 16; ++t) {
            sacc[t][i] *= temp;
            m = fmaxf(m, sacc[t][i]);
        }
        for (int d = 1; d < 16; d <<= 1) m = fmaxf(m, __shfl_xor(m, d));
        float s = 0.f;
        for (int t = 0; t < 16; ++t) {
            float e = __expf(sacc[t][i] - m);
            sacc[t][i] = e;
            s += e;
        }
        for (int d = 1; d < 16; d <<= 1) s += __shfl_xor(s, d);
        const float inv = 1.0f / s;
        for (int t = 0; t < 16; ++t) sacc[t][i] *= inv;
    }

    // ---- re-stripe P (C layout -> A layout) via wave-private LDS ----
    _Float16* pw = &lds_p[wave][0];
    for (int t = 0; t < 16; ++t)
        for (int i = 0; i < 8; ++i)
            pw[(i + hi * 8) * SEQ_K + t * 16 + lm] = (_Float16)sacc[t][i];
    __asm__ volatile("s_wait_dscnt 0" ::: "memory");

    // ---- out = P @ V : 16x64, K-dim 256 (8 wmma steps x 4 tiles) ----
    v8f oacc[4];
    for (int t = 0; t < 4; ++t) oacc[t] = zacc;
    for (int c = 0; c < 8; ++c) {
        const _Float16* pr = &pw[lm * SEQ_K + c * 32];
        v8h lo = *(const v8h*)(pr + hi * 16);
        v8h hv = *(const v8h*)(pr + 8 + hi * 16);
        v16h af;
        for (int j = 0; j < 8; ++j) { af[j] = lo[j]; af[8 + j] = hv[j]; }
        for (int t = 0; t < 4; ++t) {
            v16h bf = *(const v16h*)(&lds_vt[(t * 16 + lm) * SEQ_K + c * 32 + hi * 16]);
            oacc[t] = __builtin_amdgcn_wmma_f32_16x16x32_f16(
                false, af, false, bf, (short)0, oacc[t], false, false);
        }
    }

    for (int t = 0; t < 4; ++t)
        for (int i = 0; i < 8; ++i) {
            const size_t row = (size_t)(b * SEQ_N + n0 + i + hi * 8);
            O[row * DIMC + h * HDIM + t * 16 + lm] = (_Float16)oacc[t][i];
        }
}

// ---------------------------------------------------------------------------
// host-side orchestration
// ---------------------------------------------------------------------------
extern "C" void kernel_launch(void* const* d_in, const int* in_sizes, int n_in,
                              void* d_out, int out_size, void* d_ws, size_t ws_size,
                              hipStream_t stream) {
    const float* X    = (const float*)d_in[0];
    const float* S    = (const float*)d_in[1];
    const float* Wq   = (const float*)d_in[2];
    const float* Wk   = (const float*)d_in[3];
    const float* Wv   = (const float*)d_in[4];
    const float* Wo   = (const float*)d_in[5];
    const float* temp = (const float*)d_in[6];
    float*       out  = (float*)d_out;

    const int NX = BATCH * SEQ_N * DIMC;   // 16,777,216
    const int NS = BATCH * SEQ_K * DIMC;   //  2,097,152 f32 elems? no: 8*256*512 = 1,048,576
    const int NW = DIMC * DIMC;            //    262,144

    _Float16* ws  = (_Float16*)d_ws;
    size_t o = 0;
    _Float16* Xh  = ws + o; o += (size_t)NX;
    _Float16* Sh  = ws + o; o += (size_t)BATCH * SEQ_K * DIMC;
    _Float16* Wqh = ws + o; o += NW;
    _Float16* Wkh = ws + o; o += NW;
    _Float16* Wvh = ws + o; o += NW;
    _Float16* Woh = ws + o; o += NW;
    _Float16* Qh  = ws + o; o += (size_t)NX;
    _Float16* Kph = ws + o; o += (size_t)BATCH * SEQ_K * DIMC;
    _Float16* Vph = ws + o; o += (size_t)BATCH * SEQ_K * DIMC;
    _Float16* Oh  = ws + o;

    const int NSr = BATCH * SEQ_K * DIMC;

    cvt_f32_to_f16<<<8192, 256, 0, stream>>>(X, Xh, NX);
    cvt_f32_to_f16<<<1024, 256, 0, stream>>>(S, Sh, NSr);
    cvt_f32_to_f16<<<256, 256, 0, stream>>>(Wq, Wqh, NW);
    cvt_f32_to_f16<<<256, 256, 0, stream>>>(Wk, Wkh, NW);
    cvt_f32_to_f16<<<256, 256, 0, stream>>>(Wv, Wvh, NW);
    cvt_f32_to_f16<<<256, 256, 0, stream>>>(Wo, Woh, NW);

    // projections: Q = X @ Wq^T ; K = S @ Wk^T ; V = S @ Wv^T
    gemm_nt_wmma<false><<<dim3((BATCH * SEQ_N) / 16, DIMC / 128), 32, 0, stream>>>(
        Xh, Wqh, Qh, nullptr, nullptr);
    gemm_nt_wmma<false><<<dim3((BATCH * SEQ_K) / 16, DIMC / 128), 32, 0, stream>>>(
        Sh, Wkh, Kph, nullptr, nullptr);
    gemm_nt_wmma<false><<<dim3((BATCH * SEQ_K) / 16, DIMC / 128), 32, 0, stream>>>(
        Sh, Wvh, Vph, nullptr, nullptr);

    // attention
    attn_wmma<<<dim3(SEQ_N / 64, NHEAD, BATCH), 128, 0, stream>>>(
        Qh, Kph, Vph, temp, Oh);

    // output projection + residual: out = X + O @ Wo^T
    gemm_nt_wmma<true><<<dim3((BATCH * SEQ_N) / 16, DIMC / 128), 32, 0, stream>>>(
        Oh, Woh, nullptr, X, out);
}
